// XPUInt4Linear_78898549227775
// MI455X (gfx1250) — matrix operand order repaired and checked by reading, not compiled, not run on hardware
//
#include <hip/hip_runtime.h>
#include <cstdint>

#define K_DIM 4096
#define N_DIM 4096
#define PACKED_COLS 512          // K_DIM/8 int32 per output row

typedef __attribute__((ext_vector_type(16))) __bf16 v16bf;
typedef __attribute__((ext_vector_type(8)))  __bf16 v8bf;
typedef __attribute__((ext_vector_type(8)))  float  v8f;
typedef __attribute__((ext_vector_type(8)))  unsigned short us8;

__device__ __forceinline__ unsigned short f32_bf16_rne(float f) {
    unsigned int u = __float_as_uint(f);
    u += 0x7FFFu + ((u >> 16) & 1u);   // round-to-nearest-even
    return (unsigned short)(u >> 16);
}

// ---------------------------------------------------------------------------
// Pre-pass 1: x fp32 -> bf16 (one pass, bandwidth-bound)
// ---------------------------------------------------------------------------
__global__ __launch_bounds__(256) void convert_x_bf16(
    const float* __restrict__ x, unsigned short* __restrict__ xb, long long n)
{
    long long i = ((long long)blockIdx.x * 256 + threadIdx.x) * 8;
    if (i + 8 > n) return;
    const float4 a0 = *(const float4*)(x + i);
    const float4 a1 = *(const float4*)(x + i + 4);
    us8 v;
    v[0] = f32_bf16_rne(a0.x); v[1] = f32_bf16_rne(a0.y);
    v[2] = f32_bf16_rne(a0.z); v[3] = f32_bf16_rne(a0.w);
    v[4] = f32_bf16_rne(a1.x); v[5] = f32_bf16_rne(a1.y);
    v[6] = f32_bf16_rne(a1.z); v[7] = f32_bf16_rne(a1.w);
    *(us8*)(xb + i) = v;
}

// ---------------------------------------------------------------------------
// Pre-pass 2: int4 W -> bf16 [N][K] (magic-number dequant, one pass)
// ---------------------------------------------------------------------------
__global__ __launch_bounds__(256) void dequant_w_bf16(
    const int* __restrict__ qw, const float* __restrict__ qs,
    const int* __restrict__ qz, unsigned short* __restrict__ wb)
{
    const int idx = blockIdx.x * 256 + threadIdx.x;   // 0 .. N*512-1
    const int n = idx >> 9;
    const int c = idx & 511;
    const int g = c >> 4;                             // (c*8)/128
    const float s = qs[g * N_DIM + n];
    const float b = ((float)qz[g * N_DIM + n] + 8388608.0f) * s;
    const int q = qw[idx];
    us8 v;
    #pragma unroll
    for (int j = 0; j < 8; ++j) {
        float m = __uint_as_float(0x4B000000u | (unsigned)((q >> (4 * j)) & 15));
        v[j] = f32_bf16_rne(fmaf(m, s, -b));
    }
    *(us8*)(wb + (long long)idx * 8) = v;
}

// ---------------------------------------------------------------------------
// Async global -> LDS copy of one 64-byte row segment (4 x B128).
// INST_OFFSET adds to BOTH the LDS and global addresses (ISA 10.x), so one
// LDS-addr VGPR + one 64-bit global addr covers all four chunks.
// ---------------------------------------------------------------------------
__device__ __forceinline__ void async_copy64(unsigned lds_byte,
                                             const unsigned short* g)
{
    unsigned long long ga = (unsigned long long)(uintptr_t)g;
    asm volatile("global_load_async_to_lds_b128 %0, %1, off"
                 :: "v"(lds_byte), "v"(ga) : "memory");
    asm volatile("global_load_async_to_lds_b128 %0, %1, off offset:16"
                 :: "v"(lds_byte), "v"(ga) : "memory");
    asm volatile("global_load_async_to_lds_b128 %0, %1, off offset:32"
                 :: "v"(lds_byte), "v"(ga) : "memory");
    asm volatile("global_load_async_to_lds_b128 %0, %1, off offset:48"
                 :: "v"(lds_byte), "v"(ga) : "memory");
}

// ---------------------------------------------------------------------------
// Main GEMM: y[M][N] = xb[M][K] * wb[N][K]^T
// Double-buffered LDS, async global->LDS staging, bf16 WMMA compute.
// ---------------------------------------------------------------------------
#define BM 128
#define BN 128
#define BK 64
#define LSTR 72        // BK + 8 pad (bf16 elems); row stride 144 B

__global__ __launch_bounds__(256) void gemm_bf16_wmma(
    const unsigned short* __restrict__ A, const unsigned short* __restrict__ B,
    float* __restrict__ y, int M)
{
    __shared__ __align__(16) unsigned short As[2][BM * LSTR];   // 2 x 18 KB
    __shared__ __align__(16) unsigned short Bs[2][BN * LSTR];   // 2 x 18 KB

    const int tid  = threadIdx.x;
    const int lane = tid & 31;
    const int wave = tid >> 5;                 // 0..7
    const int m0 = blockIdx.y * BM;
    const int n0 = blockIdx.x * BN;

    const int m_wave = (wave >> 1) * 32;       // 4 waves down M
    const int n_wave = (wave & 1) * 64;        // 2 waves across N
    const int hi  = lane >> 4;
    const int l16 = lane & 15;

    // staging: each thread owns one 64B row-segment (32 bf16)
    const int srow  = tid >> 1;                // 0..127
    const int shalf = (tid & 1) * 32;          // 0 or 32 elems

    const unsigned short* agp = A + (long long)(m0 + srow) * K_DIM + shalf;
    const unsigned short* bgp = B + (long long)(n0 + srow) * K_DIM + shalf;
    const unsigned aoff0 = (unsigned)(uintptr_t)&As[0][srow * LSTR + shalf];
    const unsigned aoff1 = (unsigned)(uintptr_t)&As[1][srow * LSTR + shalf];
    const unsigned boff0 = (unsigned)(uintptr_t)&Bs[0][srow * LSTR + shalf];
    const unsigned boff1 = (unsigned)(uintptr_t)&Bs[1][srow * LSTR + shalf];

    v8f acc[2][4] = {};

    // prologue: stage tile 0 into buffer 0
    async_copy64(aoff0, agp);
    async_copy64(boff0, bgp);
    asm volatile("s_wait_asynccnt 0x0" ::: "memory");
    __syncthreads();

    for (int k0 = 0; k0 < K_DIM; k0 += BK) {
        const int cur = (k0 >> 6) & 1;
        const unsigned short* as = As[cur];
        const unsigned short* bs = Bs[cur];

        // issue async staging for next tile into the other buffer
        if (k0 + BK < K_DIM) {
            async_copy64(cur ? aoff0 : aoff1, agp + k0 + BK);
            async_copy64(cur ? boff0 : boff1, bgp + k0 + BK);
        }

        #pragma unroll
        for (int kk = 0; kk < BK; kk += 32) {
            // A fragments (ISA 16-bit A 16x32 layout:
            //   lanes 0-15: K{0..7,16..23}, lanes 16-31: K{8..15,24..31})
            v16bf af[2];
            #pragma unroll
            for (int mt = 0; mt < 2; ++mt) {
                const unsigned short* abase =
                    &as[(m_wave + mt * 16 + l16) * LSTR + kk];
                v8bf a_lo = *(const v8bf*)(abase + hi * 8);
                v8bf a_hi = *(const v8bf*)(abase + 16 + hi * 8);
                af[mt] = __builtin_shufflevector(a_lo, a_hi,
                    0,1,2,3,4,5,6,7,8,9,10,11,12,13,14,15);
            }
            // B fragments (lanes 0-15: K{0..7,16..23}, lanes 16-31: K{8..15,24..31})
            v16bf bfv[4];
            #pragma unroll
            for (int nt = 0; nt < 4; ++nt) {
                const unsigned short* bbase =
                    &bs[(n_wave + nt * 16 + l16) * LSTR + kk];
                v8bf b_lo = *(const v8bf*)(bbase + hi * 8);
                v8bf b_hi = *(const v8bf*)(bbase + 16 + hi * 8);
                bfv[nt] = __builtin_shufflevector(b_lo, b_hi,
                    0,1,2,3,4,5,6,7,8,9,10,11,12,13,14,15);
            }
            #pragma unroll
            for (int nt = 0; nt < 4; ++nt)
                #pragma unroll
                for (int mt = 0; mt < 2; ++mt)
                    acc[mt][nt] = __builtin_amdgcn_wmma_f32_16x16x32_bf16(
                        false, af[mt], false, bfv[nt], (short)0, acc[mt][nt],
                        false, false);
        }

        // make next buffer visible to all waves, then one barrier per iter
        if (k0 + BK < K_DIM)
            asm volatile("s_wait_asynccnt 0x0" ::: "memory");
        __syncthreads();
    }

    // epilogue: 16x16 f32 C layout (VGPR r -> M=r / M=8+r, lane -> N)
    #pragma unroll
    for (int mt = 0; mt < 2; ++mt) {
        const int om = m0 + m_wave + mt * 16 + hi * 8;
        #pragma unroll
        for (int nt = 0; nt < 4; ++nt) {
            const int on = n0 + n_wave + nt * 16 + l16;
            #pragma unroll
            for (int r = 0; r < 8; ++r)
                y[(long long)(om + r) * N_DIM + on] = acc[mt][nt][r];
        }
    }
}

// ---------------------------------------------------------------------------
// Fallback fused kernel (used only when d_ws is too small)
// ---------------------------------------------------------------------------
#define FBM 64
#define FBN 128
#define FBK 32
#define FLSTR 40

__global__ __launch_bounds__(256) void int4_gemm_fused(
    const float* __restrict__ x, const int* __restrict__ qweight,
    const float* __restrict__ qscale, const int* __restrict__ qzeros,
    float* __restrict__ y, int M)
{
    __shared__ __align__(16) unsigned short As[FBM * FLSTR];
    __shared__ __align__(16) unsigned short Bs[FBN * FLSTR];

    const int tid  = threadIdx.x;
    const int lane = tid & 31;
    const int wave = tid >> 5;
    const int m0 = blockIdx.y * FBM;
    const int n0 = blockIdx.x * FBN;
    const int m_wave = (wave >> 1) * 16;
    const int n_wave = (wave & 1) * 64;
    const int hi  = lane >> 4;
    const int l16 = lane & 15;
    const int arow = tid >> 2;
    const int acol = (tid & 3) * 8;
    const int brow  = tid >> 1;
    const int bhalf = tid & 1;

    v8f acc[4] = {};
    const float* xrow = x + (long long)(m0 + arow) * K_DIM + acol;
    const int*   qrow = qweight + (long long)(n0 + brow) * PACKED_COLS + bhalf * 2;
    const int    ncol = n0 + brow;

    for (int k0 = 0; k0 < K_DIM; k0 += FBK) {
        const float4 a0 = *(const float4*)(xrow + k0);
        const float4 a1 = *(const float4*)(xrow + k0 + 4);
        us8 av;
        av[0] = f32_bf16_rne(a0.x); av[1] = f32_bf16_rne(a0.y);
        av[2] = f32_bf16_rne(a0.z); av[3] = f32_bf16_rne(a0.w);
        av[4] = f32_bf16_rne(a1.x); av[5] = f32_bf16_rne(a1.y);
        av[6] = f32_bf16_rne(a1.z); av[7] = f32_bf16_rne(a1.w);
        *(us8*)&As[arow * FLSTR + acol] = av;

        const int   g  = k0 >> 7;
        const float sc = qscale[g * N_DIM + ncol];
        const float b  = ((float)qzeros[g * N_DIM + ncol] + 8388608.0f) * sc;
        const int q0 = qrow[(k0 >> 3) + 0];
        const int q1 = qrow[(k0 >> 3) + 1];
        us8 b0v, b1v;
        #pragma unroll
        for (int j = 0; j < 8; ++j) {
            float m0f = __uint_as_float(0x4B000000u | (unsigned)((q0 >> (4 * j)) & 15));
            float m1f = __uint_as_float(0x4B000000u | (unsigned)((q1 >> (4 * j)) & 15));
            b0v[j] = f32_bf16_rne(fmaf(m0f, sc, -b));
            b1v[j] = f32_bf16_rne(fmaf(m1f, sc, -b));
        }
        unsigned short* bdst = &Bs[brow * FLSTR + bhalf * 16];
        *(us8*)(bdst)     = b0v;
        *(us8*)(bdst + 8) = b1v;

        __syncthreads();

        const unsigned short* abase = &As[(m_wave + l16) * FLSTR];
        v8bf a_lo = *(const v8bf*)(abase + hi * 8);
        v8bf a_hi = *(const v8bf*)(abase + 16 + hi * 8);
        v16bf afrag = __builtin_shufflevector(a_lo, a_hi,
            0,1,2,3,4,5,6,7,8,9,10,11,12,13,14,15);

        #pragma unroll
        for (int t = 0; t < 4; ++t) {
            const unsigned short* bbase =
                &Bs[(n_wave + t * 16 + l16) * FLSTR];
            v8bf b_lo = *(const v8bf*)(bbase + hi * 8);
            v8bf b_hi = *(const v8bf*)(bbase + 16 + hi * 8);
            v16bf bfrag = __builtin_shufflevector(b_lo, b_hi,
                0,1,2,3,4,5,6,7,8,9,10,11,12,13,14,15);
            acc[t] = __builtin_amdgcn_wmma_f32_16x16x32_bf16(
                false, afrag, false, bfrag, (short)0, acc[t], false, false);
        }
        __syncthreads();
    }

    const int om = m0 + m_wave + hi * 8;
    const int on = n0 + n_wave + l16;
    #pragma unroll
    for (int t = 0; t < 4; ++t)
        #pragma unroll
        for (int r = 0; r < 8; ++r)
            y[(long long)(om + r) * N_DIM + on + t * 16] = acc[t][r];
}

// ---------------------------------------------------------------------------
extern "C" void kernel_launch(void* const* d_in, const int* in_sizes, int n_in,
                              void* d_out, int out_size, void* d_ws, size_t ws_size,
                              hipStream_t stream) {
    const float* x       = (const float*)d_in[0];
    const int*   qweight = (const int*)d_in[1];
    const float* qscale  = (const float*)d_in[2];
    const int*   qzeros  = (const int*)d_in[3];
    float* y = (float*)d_out;

    const long long xElems = (long long)in_sizes[0];      // M*K
    const int M = (int)(xElems / K_DIM);                  // 8192
    const long long wElems = (long long)N_DIM * K_DIM;
    const size_t need = (size_t)(xElems + wElems) * 2;    // ~96 MB

    if (ws_size >= need) {
        unsigned short* xb = (unsigned short*)d_ws;
        unsigned short* wb = xb + xElems;
        convert_x_bf16<<<(unsigned)((xElems / 8 + 255) / 256), 256, 0, stream>>>(
            x, xb, xElems);
        dequant_w_bf16<<<(N_DIM * PACKED_COLS) / 256, 256, 0, stream>>>(
            qweight, qscale, qzeros, wb);
        dim3 grid(N_DIM / BN, M / BM);
        gemm_bf16_wmma<<<grid, dim3(256), 0, stream>>>(xb, wb, y, M);
    } else {
        dim3 grid(N_DIM / FBN, M / FBM);
        int4_gemm_fused<<<grid, dim3(256), 0, stream>>>(
            x, qweight, qscale, qzeros, y, M);
    }
}
